// EGCNNet_41721312313871
// MI455X (gfx1250) — matrix operand order repaired and checked by reading, not compiled, not run on hardware
//
#include <hip/hip_runtime.h>
#include <hip/hip_bf16.h>

// ---------------------------------------------------------------------------
// EGCNNet on gfx1250 (MI455X): WMMA bf16 GEMMs, async-to-LDS weight staging,
// L2-resident gather/scatter.
// ---------------------------------------------------------------------------

typedef __bf16 v16bf __attribute__((ext_vector_type(16)));
typedef float  v8f   __attribute__((ext_vector_type(8)));

#define EPS 1e-5f

__device__ __forceinline__ __bf16 f2bf(float f) {
  union { float f; unsigned u; } x; x.f = f;
  unsigned r = x.u + 0x7FFFu + ((x.u >> 16) & 1u);   // round-to-nearest-even
  unsigned short hs = (unsigned short)(r >> 16);
  __bf16 o; __builtin_memcpy(&o, &hs, 2); return o;
}

__device__ __forceinline__ v8f v8f_zero() {
  v8f z;
#pragma unroll
  for (int i = 0; i < 8; ++i) z[i] = 0.0f;
  return z;
}

__device__ __forceinline__ v8f bwmma(v16bf a, v16bf b, v8f c) {
  return __builtin_amdgcn_wmma_f32_16x16x32_bf16(false, a, false, b,
                                                 (short)0, c, false, false);
}

// Stage `bytes` of packed weights (16B-aligned) from global into LDS using the
// CDNA5 async copy path (GLOBAL_LOAD_ASYNC_TO_LDS_B128, tracked on ASYNCcnt).
// Each wave waits for its own async transfers, then the workgroup barrier
// makes the full block visible. Must be called by ALL threads of the block.
__device__ __forceinline__ void stage_async(const __bf16* __restrict__ gsrc,
                                            __bf16* smem, int bytes) {
  unsigned ldsOff = (unsigned)(uintptr_t)smem;   // low 32 bits = LDS offset
  for (int base = threadIdx.x * 16; base < bytes; base += 256 * 16) {
    unsigned d = ldsOff + (unsigned)base;
    const char* g = (const char*)gsrc + base;
    asm volatile("global_load_async_to_lds_b128 %0, %1, off"
                 :: "v"(d), "v"(g) : "memory");
  }
  asm volatile("s_wait_asynccnt 0x0" ::: "memory");
  __syncthreads();
}

// ----------------------------- utility kernels ------------------------------

__global__ void k_zero(float* __restrict__ p, long n) {
  long i = (long)blockIdx.x * blockDim.x + threadIdx.x;
  long st = (long)gridDim.x * blockDim.x;
  for (; i < n; i += st) p[i] = 0.0f;
}

__global__ void k_relu(float* __restrict__ p, long n) {
  long i = (long)blockIdx.x * blockDim.x + threadIdx.x;
  long st = (long)gridDim.x * blockDim.x;
  for (; i < n; i += st) p[i] = fmaxf(p[i], 0.0f);
}

__global__ void k_deg_count(const int* __restrict__ dst, float* __restrict__ deg, int E) {
  long i = (long)blockIdx.x * blockDim.x + threadIdx.x;
  long st = (long)gridDim.x * blockDim.x;
  for (; i < E; i += st) atomicAdd(&deg[dst[i]], 1.0f);
}

__global__ void k_dinv(float* __restrict__ deg, int N) {
  long i = (long)blockIdx.x * blockDim.x + threadIdx.x;
  long st = (long)gridDim.x * blockDim.x;
  for (; i < N; i += st) deg[i] = rsqrtf(fmaxf(deg[i], 1.0f));
}

__global__ void k_edge_norm(const float* __restrict__ dinv, const int* __restrict__ src,
                            const int* __restrict__ dst, float* __restrict__ nrm, int E) {
  long i = (long)blockIdx.x * blockDim.x + threadIdx.x;
  long st = (long)gridDim.x * blockDim.x;
  for (; i < E; i += st) nrm[i] = dinv[src[i]] * dinv[dst[i]];
}

// Column sums / sums-of-squares with intra-block reduction (Kc = 32 or 64).
__global__ __launch_bounds__(256) void k_colstats(const float* __restrict__ X, int M, int Kc,
                                                  float* __restrict__ sums,
                                                  float* __restrict__ sumsq) {
  __shared__ float sh[256], sh2[256];
  const int tid = threadIdx.x;
  const int col = tid % Kc;
  const int rsub = tid / Kc;
  const int rowsPerPass = 256 / Kc;
  float s = 0.0f, s2 = 0.0f;
  for (long r = (long)blockIdx.x * rowsPerPass + rsub; r < M;
       r += (long)gridDim.x * rowsPerPass) {
    float v = X[r * Kc + col];
    s += v; s2 += v * v;
  }
  sh[tid] = s; sh2[tid] = s2;
  __syncthreads();
  for (int off = 128; off >= Kc; off >>= 1) {
    if (tid < off) { sh[tid] += sh[tid + off]; sh2[tid] += sh2[tid + off]; }
    __syncthreads();
  }
  if (tid < Kc) { atomicAdd(&sums[tid], sh[tid]); atomicAdd(&sumsq[tid], sh2[tid]); }
}

// a = w * rsqrt(var+eps); c = b - mean * a   (BN folding coefficients)
__global__ void k_bn_fold(const float* __restrict__ sums, const float* __restrict__ sumsq,
                          float invM, const float* __restrict__ w, const float* __restrict__ b,
                          float* __restrict__ a, float* __restrict__ c, int Kc) {
  int t = threadIdx.x;
  if (t >= Kc) return;
  float mean = sums[t] * invM;
  float var = sumsq[t] * invM - mean * mean;
  float s = rsqrtf(var + EPS);
  float av = w[t] * s;
  a[t] = av;
  c[t] = b[t] - mean * av;
}

// bfold[n] = sum_k c[k]*W[k,n] + b0[n]
__global__ void k_fold_bias(const float* __restrict__ W, int K, const float* __restrict__ c,
                            const float* __restrict__ b0, float* __restrict__ bf) {
  int n = threadIdx.x;
  if (n >= 64) return;
  float s = b0 ? b0[n] : 0.0f;
  for (int k = 0; k < K; ++k) s += c[k] * W[k * 64 + n];
  bf[n] = s;
}

// Pack W[K,64] (optionally row-scaled by a[k], zero-padded to K=64) into the
// WMMA-B register layout: element ((ks*4+nt)*32+lane)*16+hh holds
// W[ks*32 + (lane>>4)*8 + (hh&7) + (hh>>3)*16][nt*16 + (lane&15)] as bf16.
__global__ void k_pack_b(const float* __restrict__ W, const float* __restrict__ scale,
                         __bf16* __restrict__ Wpk, int K) {
  int idx = blockIdx.x * blockDim.x + threadIdx.x;
  if (idx >= 4096) return;
  int hh = idx & 15;
  int lane = (idx >> 4) & 31;
  int nt = (idx >> 9) & 3;
  int ks = idx >> 11;
  int k = ks * 32 + ((lane >> 4) << 3) + (hh & 7) + ((hh >> 3) << 4);
  int n = nt * 16 + (lane & 15);
  float v = 0.0f;
  if (k < K) {
    v = W[k * 64 + n];
    if (scale) v *= scale[k];
  }
  Wpk[idx] = f2bf(v);
}

// ------------------------------- WMMA GEMM ----------------------------------
// C[M,64] = A[M,NKS*32] @ Wpk(+bias). 8 waves/block, 16 rows/wave. Weights
// are async-staged to LDS once per block; K loop fully unrolled.
template <int NKS>
__global__ __launch_bounds__(256) void k_gemm(const float* __restrict__ A,
                                              const __bf16* __restrict__ Wpk,
                                              const float* __restrict__ bias,
                                              float* __restrict__ C,
                                              int M, int relu) {
  __shared__ __bf16 smemW[NKS * 2048];
  stage_async(Wpk, smemW, NKS * 4096);

  const int wave = threadIdx.x >> 5;
  const int lane = threadIdx.x & 31;
  const int rowBase = (blockIdx.x * 8 + wave) * 16;
  if (rowBase >= M) return;
  const int half = lane >> 4, l15 = lane & 15;
  const int K = NKS * 32;
  int rA = rowBase + l15;
  if (rA > M - 1) rA = M - 1;

  // Load + convert the A stripes for all K-steps first (loads overlap).
  v16bf av[NKS];
#pragma unroll
  for (int ks = 0; ks < NKS; ++ks) {
    const float* ap = A + (size_t)rA * K + ks * 32 + half * 8;
#pragma unroll
    for (int hh = 0; hh < 8; ++hh) {
      av[ks][hh]     = f2bf(ap[hh]);
      av[ks][hh + 8] = f2bf(ap[16 + hh]);
    }
  }
  v8f acc[4];
#pragma unroll
  for (int nt = 0; nt < 4; ++nt) acc[nt] = v8f_zero();
#pragma unroll
  for (int ks = 0; ks < NKS; ++ks) {
#pragma unroll
    for (int nt = 0; nt < 4; ++nt) {
      v16bf bv = *(const v16bf*)(smemW + (((ks << 2) + nt) * 32 + lane) * 16);
      acc[nt] = bwmma(av[ks], bv, acc[nt]);
    }
  }
#pragma unroll
  for (int nt = 0; nt < 4; ++nt) {
    int col = nt * 16 + l15;
    float bb = bias ? bias[col] : 0.0f;
#pragma unroll
    for (int j = 0; j < 8; ++j) {
      int r = rowBase + half * 8 + j;
      if (r < M) {
        float v = acc[nt][j] + bb;
        if (relu) v = fmaxf(v, 0.0f);
        C[(size_t)r * 64 + col] = v;
      }
    }
  }
}

// --------------------- initial edge features (bf16 out) ---------------------
// e = relu(A0[src] + B0[dst] + edge_attr @ We0[2H:] + be0), one 16-edge tile
// per wave; the K=8 attr GEMM runs as a zero-padded 16x16x32 WMMA.
__global__ __launch_bounds__(256) void k_edge_init(const float* __restrict__ eattr,
                                                   const __bf16* __restrict__ WpkE0,
                                                   const float* __restrict__ be0,
                                                   const float* __restrict__ A0,
                                                   const float* __restrict__ B0,
                                                   const int* __restrict__ src,
                                                   const int* __restrict__ dst,
                                                   __bf16* __restrict__ e, int E) {
  __shared__ __bf16 smemW[2048];       // ks = 0 tiles only
  stage_async(WpkE0, smemW, 4096);

  const int wave = threadIdx.x >> 5, lane = threadIdx.x & 31;
  const long base = ((long)blockIdx.x * 8 + wave) * 16;
  if (base >= E) return;
  const int half = lane >> 4, l15 = lane & 15;

  v16bf av;
#pragma unroll
  for (int hh = 0; hh < 16; ++hh) av[hh] = f2bf(0.0f);
  {
    long r = base + l15; if (r > (long)E - 1) r = E - 1;
    if (half == 0) {
      const float* p = eattr + r * 8;
#pragma unroll
      for (int hh = 0; hh < 8; ++hh) av[hh] = f2bf(p[hh]);  // K = 0..7
    }
  }
  v8f acc[4];
#pragma unroll
  for (int nt = 0; nt < 4; ++nt) {
    acc[nt] = v8f_zero();
    v16bf bv = *(const v16bf*)(smemW + (nt * 32 + lane) * 16);
    acc[nt] = bwmma(av, bv, acc[nt]);
  }
  int ss[8], dd[8];
#pragma unroll
  for (int j = 0; j < 8; ++j) {
    long er = base + half * 8 + j; if (er > (long)E - 1) er = E - 1;
    ss[j] = src[er]; dd[j] = dst[er];
  }
#pragma unroll
  for (int nt = 0; nt < 4; ++nt) {
    int col = nt * 16 + l15;
    float bb = be0[col];
#pragma unroll
    for (int j = 0; j < 8; ++j) {
      long er = base + half * 8 + j;
      if (er < E) {
        float v = acc[nt][j] + bb + A0[(size_t)ss[j] * 64 + col] +
                  B0[(size_t)dd[j] * 64 + col];
        e[er * 64 + col] = f2bf(fmaxf(v, 0.0f));
      }
    }
  }
}

// ------------------- gated message + scatter (per layer) --------------------
// ei = eTile@We_e (WMMA) + Web + A[src] + B[dst];
// msg = norm * ht[src] * sigmoid(ei);  hnext[dst] += msg (L2 atomics).
__global__ __launch_bounds__(256) void k_edge_message(const __bf16* __restrict__ ebuf,
                                                      const __bf16* __restrict__ WpkE,
                                                      const float* __restrict__ Web,
                                                      const float* __restrict__ An,
                                                      const float* __restrict__ Bn,
                                                      const float* __restrict__ ht,
                                                      const int* __restrict__ src,
                                                      const int* __restrict__ dst,
                                                      const float* __restrict__ nrm,
                                                      float* __restrict__ hnext, int E) {
  __shared__ __bf16 smemW[4096];
  stage_async(WpkE, smemW, 8192);

  const int wave = threadIdx.x >> 5, lane = threadIdx.x & 31;
  const long base = ((long)blockIdx.x * 8 + wave) * 16;
  if (base >= E) return;
  const int half = lane >> 4, l15 = lane & 15;

  long rE = base + l15; if (rE > (long)E - 1) rE = E - 1;
  const __bf16* ep = ebuf + rE * 64 + half * 8;
  v16bf av0, av1;
#pragma unroll
  for (int hh = 0; hh < 8; ++hh) {
    av0[hh] = ep[hh];      av0[hh + 8] = ep[16 + hh];
    av1[hh] = ep[32 + hh]; av1[hh + 8] = ep[48 + hh];
  }
  v8f acc[4];
#pragma unroll
  for (int nt = 0; nt < 4; ++nt) acc[nt] = v8f_zero();
#pragma unroll
  for (int nt = 0; nt < 4; ++nt) {
    v16bf b0 = *(const v16bf*)(smemW + ((0 * 4 + nt) * 32 + lane) * 16);
    acc[nt] = bwmma(av0, b0, acc[nt]);
    v16bf b1 = *(const v16bf*)(smemW + ((1 * 4 + nt) * 32 + lane) * 16);
    acc[nt] = bwmma(av1, b1, acc[nt]);
  }
  int ss[8], dd[8]; float nm[8];
#pragma unroll
  for (int j = 0; j < 8; ++j) {
    long er = base + half * 8 + j; if (er > (long)E - 1) er = E - 1;
    ss[j] = src[er]; dd[j] = dst[er]; nm[j] = nrm[er];
  }
#pragma unroll
  for (int nt = 0; nt < 4; ++nt) {
    int col = nt * 16 + l15;
    float wb = Web[col];
#pragma unroll
    for (int j = 0; j < 8; ++j) {
      if (base + half * 8 + j < E) {
        size_t so = (size_t)ss[j] * 64 + col;
        size_t dof = (size_t)dd[j] * 64 + col;
        float ei = acc[nt][j] + wb + An[so] + Bn[dof];
        float sg = 1.0f / (1.0f + __expf(-ei));
        atomicAdd(&hnext[dof], nm[j] * ht[so] * sg);
      }
    }
  }
}

// ------------------------------ pooling + head ------------------------------

__global__ void k_pool(const float* __restrict__ h, const int* __restrict__ batch,
                       float* __restrict__ g, int N) {
  long i = (long)blockIdx.x * blockDim.x + threadIdx.x;
  long st = (long)gridDim.x * blockDim.x;
  long total = (long)N * 64;
  for (; i < total; i += st) {
    int node = (int)(i >> 6), col = (int)(i & 63);
    atomicAdd(&g[(size_t)batch[node] * 64 + col], h[i]);
  }
}

// out[row,col] = relu?( sum_k (g[row,k]*a[k]+c[k]) * W[k,col] + b[col] )
__global__ void k_fc(const float* __restrict__ g, const float* __restrict__ a,
                     const float* __restrict__ c, const float* __restrict__ W,
                     const float* __restrict__ b, float* __restrict__ out, int relu) {
  int row = blockIdx.x, col = threadIdx.x;  // 64 threads
  float s = b[col];
  for (int k = 0; k < 64; ++k)
    s += (g[(size_t)row * 64 + k] * a[k] + c[k]) * W[k * 64 + col];
  if (relu) s = fmaxf(s, 0.0f);
  out[(size_t)row * 64 + col] = s;
}

// classifier + log_softmax over 10 classes, one block (32 threads) per row
__global__ void k_cls(const float* __restrict__ g1, const float* __restrict__ a,
                      const float* __restrict__ c, const float* __restrict__ Wcls,
                      const float* __restrict__ bcls, float* __restrict__ out) {
  __shared__ float logit[10];
  __shared__ float red[2];
  int row = blockIdx.x, t = threadIdx.x;
  if (t < 10) {
    float s = bcls[t];
    for (int k = 0; k < 64; ++k)
      s += (g1[(size_t)row * 64 + k] * a[k] + c[k]) * Wcls[k * 10 + t];
    logit[t] = s;
  }
  __syncthreads();
  if (t == 0) {
    float m = logit[0];
    for (int i = 1; i < 10; ++i) m = fmaxf(m, logit[i]);
    float se = 0.0f;
    for (int i = 0; i < 10; ++i) se += __expf(logit[i] - m);
    red[0] = m; red[1] = __logf(se);
  }
  __syncthreads();
  if (t < 10) out[(size_t)row * 10 + t] = logit[t] - red[0] - red[1];
}

// --------------------------------- driver -----------------------------------

extern "C" void kernel_launch(void* const* d_in, const int* in_sizes, int n_in,
                              void* d_out, int out_size, void* d_ws, size_t ws_size,
                              hipStream_t stream) {
  const float* x     = (const float*)d_in[0];
  const float* eattr = (const float*)d_in[1];
  const int*   eidx  = (const int*)d_in[2];
  const int*   batch = (const int*)d_in[3];
  const float* bnfw  = (const float*)d_in[4];
  const float* bnfb  = (const float*)d_in[5];
  const float* Wn0   = (const float*)d_in[6];
  const float* bn0   = (const float*)d_in[7];
  const float* We0   = (const float*)d_in[8];
  const float* be0   = (const float*)d_in[9];
  const float* bnsw  = (const float*)d_in[10];
  const float* bnsb  = (const float*)d_in[11];
  const float* Wn    = (const float*)d_in[12];
  const float* Wnb   = (const float*)d_in[13];
  const float* We    = (const float*)d_in[14];
  const float* Web   = (const float*)d_in[15];
  const float* bnfcw = (const float*)d_in[16];
  const float* bnfcb = (const float*)d_in[17];
  const float* Wfc   = (const float*)d_in[18];
  const float* bfc   = (const float*)d_in[19];
  const float* bnhw  = (const float*)d_in[20];
  const float* bnhb  = (const float*)d_in[21];
  const float* Wcls  = (const float*)d_in[22];
  const float* bcls  = (const float*)d_in[23];
  float* out = (float*)d_out;

  const int N = in_sizes[0] / 32;
  const int E = in_sizes[1] / 8;
  const int G = 512;
  const int* srcI = eidx;
  const int* dstI = eidx + E;

  // ---- workspace carving (256B aligned) ----
  char* ws = (char*)d_ws;
  size_t off = 0;
  auto carve = [&](size_t bytes) -> char* {
    char* p = ws + off;
    off += (bytes + 255) & ~(size_t)255;
    return p;
  };
  float*  deg   = (float*)carve((size_t)N * 4);
  float*  nrm   = (float*)carve((size_t)E * 4);
  float*  h     = (float*)carve((size_t)N * 64 * 4);   // also serves as hnext
  float*  ht    = (float*)carve((size_t)N * 64 * 4);
  float*  An    = (float*)carve((size_t)N * 64 * 4);
  float*  Bn    = (float*)carve((size_t)N * 64 * 4);
  __bf16* ebuf  = (__bf16*)carve((size_t)E * 64 * 2);
  float*  g     = (float*)carve((size_t)G * 64 * 4);
  float*  g1    = (float*)carve((size_t)G * 64 * 4);
  float*  sums  = (float*)carve(128 * 4);
  float*  sumsq = sums + 64;
  float*  avec  = (float*)carve(64 * 4);
  float*  cvec  = (float*)carve(64 * 4);
  float*  bfold = (float*)carve(64 * 4);
  __bf16* WpkM  = (__bf16*)carve(4096 * 2);
  __bf16* WpkA  = (__bf16*)carve(4096 * 2);
  __bf16* WpkB  = (__bf16*)carve(4096 * 2);
  __bf16* WpkE  = (__bf16*)carve(4096 * 2);
  (void)n_in; (void)out_size; (void)ws_size;

  const int TB = 256;
  const int gs = 1024;                         // grid-stride blocks
  const int gemmGrid = (N + 127) / 128;        // 8 waves x 16 rows per block
  const int edgeGrid = (E + 127) / 128;        // 8 tiles of 16 edges per block
  const int statGrid = 512;

  // ---- degree / symmetric edge norm ----
  k_zero<<<gs, TB, 0, stream>>>(deg, N);
  k_deg_count<<<gs, TB, 0, stream>>>(dstI, deg, E);
  k_dinv<<<gs, TB, 0, stream>>>(deg, N);
  k_edge_norm<<<gs, TB, 0, stream>>>(deg, srcI, dstI, nrm, E);

  // ---- feature conv: h0 = BN(x)@Wn0 + bn0 (pre-relu kept for e) ----
  k_zero<<<1, 128, 0, stream>>>(sums, 128);
  k_colstats<<<statGrid, TB, 0, stream>>>(x, N, 32, sums, sumsq);
  k_bn_fold<<<1, 64, 0, stream>>>(sums, sumsq, 1.0f / (float)N, bnfw, bnfb, avec, cvec, 32);
  k_pack_b<<<16, 256, 0, stream>>>(Wn0, avec, WpkM, 32);
  k_fold_bias<<<1, 64, 0, stream>>>(Wn0, 32, cvec, bn0, bfold);
  k_gemm<1><<<gemmGrid, TB, 0, stream>>>(x, WpkM, bfold, h, N, 0);

  // ---- e = relu(h0[src]@We0a + h0[dst]@We0b + ea@We0e + be0), bf16 ----
  k_pack_b<<<16, 256, 0, stream>>>(We0, nullptr, WpkA, 64);
  k_pack_b<<<16, 256, 0, stream>>>(We0 + 64 * 64, nullptr, WpkB, 64);
  k_pack_b<<<16, 256, 0, stream>>>(We0 + 128 * 64, nullptr, WpkE, 8);
  k_gemm<2><<<gemmGrid, TB, 0, stream>>>(h, WpkA, nullptr, An, N, 0);
  k_gemm<2><<<gemmGrid, TB, 0, stream>>>(h, WpkB, nullptr, Bn, N, 0);
  k_edge_init<<<edgeGrid, TB, 0, stream>>>(eattr, WpkE, be0, An, Bn, srcI, dstI, ebuf, E);
  k_relu<<<gs, TB, 0, stream>>>(h, (long)N * 64);

  // ---- 3 edge-gated GCN layers ----
  for (int i = 0; i < 3; ++i) {
    k_zero<<<1, 128, 0, stream>>>(sums, 128);
    k_colstats<<<statGrid, TB, 0, stream>>>(h, N, 64, sums, sumsq);
    k_bn_fold<<<1, 64, 0, stream>>>(sums, sumsq, 1.0f / (float)N,
                                    bnsw + i * 64, bnsb + i * 64, avec, cvec, 64);
    k_pack_b<<<16, 256, 0, stream>>>(Wn + i * 4096, avec, WpkM, 64);
    k_fold_bias<<<1, 64, 0, stream>>>(Wn + i * 4096, 64, cvec, Wnb + i * 64, bfold);
    k_gemm<2><<<gemmGrid, TB, 0, stream>>>(h, WpkM, bfold, ht, N, 0);

    const float* Wei = We + i * 12288;
    k_pack_b<<<16, 256, 0, stream>>>(Wei, nullptr, WpkA, 64);
    k_pack_b<<<16, 256, 0, stream>>>(Wei + 4096, nullptr, WpkB, 64);
    k_pack_b<<<16, 256, 0, stream>>>(Wei + 8192, nullptr, WpkE, 64);
    k_gemm<2><<<gemmGrid, TB, 0, stream>>>(ht, WpkA, nullptr, An, N, 0);
    k_gemm<2><<<gemmGrid, TB, 0, stream>>>(ht, WpkB, nullptr, Bn, N, 0);

    k_zero<<<gs, TB, 0, stream>>>(h, (long)N * 64);   // h becomes hnext
    k_edge_message<<<edgeGrid, TB, 0, stream>>>(ebuf, WpkE, Web + i * 64,
                                                An, Bn, ht, srcI, dstI, nrm, h, E);
    k_relu<<<gs, TB, 0, stream>>>(h, (long)N * 64);
  }

  // ---- global_add_pool + classifier head ----
  k_zero<<<gs, TB, 0, stream>>>(g, (long)G * 64);
  k_pool<<<gs, TB, 0, stream>>>(h, batch, g, N);

  k_zero<<<1, 128, 0, stream>>>(sums, 128);
  k_colstats<<<64, TB, 0, stream>>>(g, G, 64, sums, sumsq);
  k_bn_fold<<<1, 64, 0, stream>>>(sums, sumsq, 1.0f / (float)G, bnfcw, bnfcb, avec, cvec, 64);
  k_fc<<<G, 64, 0, stream>>>(g, avec, cvec, Wfc, bfc, g1, 1);

  k_zero<<<1, 128, 0, stream>>>(sums, 128);
  k_colstats<<<64, TB, 0, stream>>>(g1, G, 64, sums, sumsq);
  k_bn_fold<<<1, 64, 0, stream>>>(sums, sumsq, 1.0f / (float)G, bnhw, bnhb, avec, cvec, 64);
  k_cls<<<G, 32, 0, stream>>>(g1, avec, cvec, Wcls, bcls, out);
}
